// RNN_25194278159102
// MI455X (gfx1250) — compile-verified, matrix operand order checked by reading, not run on hardware
//
#include <hip/hip_runtime.h>

typedef __attribute__((ext_vector_type(16))) _Float16 v16h;
typedef __attribute__((ext_vector_type(8)))  float    v8f;

#define HID 24

static __device__ __forceinline__ v8f wmma_f16(v16h a, v16h b, v8f c) {
  // D = A(16x32 f16) x B(32x16 f16) + C(16x16 f32)
  return __builtin_amdgcn_wmma_f32_16x16x32_f16(
      /*neg_a=*/false, a, /*neg_b=*/false, b,
      /*c_mod=*/(short)0, c, /*reuse_a=*/false, /*reuse_b=*/false);
}

// Convert two f32 C-tiles to one f16 B operand, relu applied in packed f16.
// (cvt is monotonic and cvt(0)==0, so max(cvt(x),0) == cvt(max(x,0)).)
static __device__ __forceinline__ v16h pack_relu(v8f p0, v8f p1) {
  v16h nb;
#pragma unroll
  for (int r = 0; r < 8; ++r) {
    nb[r]     = (_Float16)p0[r];
    nb[8 + r] = (_Float16)p1[r];
  }
  v16h zv = {};
#if __has_builtin(__builtin_elementwise_max)
  nb = __builtin_elementwise_max(nb, zv);       // v_pk_max_num_f16
#else
#pragma unroll
  for (int e = 0; e < 16; ++e) nb[e] = (nb[e] > (_Float16)0) ? nb[e] : (_Float16)0;
#endif
  return nb;
}

__global__ __launch_bounds__(32)
void rnn3_wmma_kernel(const float* __restrict__ x,        // [B,T]
                      const float* __restrict__ hin,      // [3,B,24]
                      const float* __restrict__ Wih0, const float* __restrict__ bih0,
                      const float* __restrict__ Whh0, const float* __restrict__ bhh0,
                      const float* __restrict__ Wih1, const float* __restrict__ bih1,
                      const float* __restrict__ Whh1, const float* __restrict__ bhh1,
                      const float* __restrict__ Wih2, const float* __restrict__ bih2,
                      const float* __restrict__ Whh2, const float* __restrict__ bhh2,
                      const float* __restrict__ W1,  const float* __restrict__ b1,
                      const float* __restrict__ W2,  const float* __restrict__ b2,
                      float* __restrict__ out, int B, int T)
{
  const int lane = threadIdx.x;      // 0..31 (wave32)
  const int col  = lane & 15;        // batch column within tile
  const bool hi  = lane >= 16;
  const int bbase = blockIdx.x * 16; // batch base for this wave

  // ---------- constant A operands (weights, transposed formulation) ----------
  // A tile t covers output rows m = 16t .. 16t+15 (rows >= 24 zero padded).
  // f16 A layout per lane: e<8 -> K = (hi?8:0)+e ; e>=8 -> K = 16+(hi?8:0)+(e-8)
  auto buildA = [&](const float* W, const float* aug, int tile) -> v16h {
    int row = tile * 16 + col;
    int kb  = hi ? 8 : 0;
    v16h a;
#pragma unroll
    for (int e = 0; e < 16; ++e) {
      int k = (e < 8) ? (kb + e) : (16 + kb + (e - 8));
      float v = 0.0f;
      if (row < HID) {
        if (k < HID)              v = W[row * HID + k];
        else if (k == HID && aug) v = aug[row];      // augmented column (W_ih0)
      }
      a[e] = (_Float16)v;
    }
    return a;
  };

  // C-init = combined bias, broadcast over batch columns.
  // C layout: VGPR r -> row m = 16*tile + r + (hi?8:0)
  auto buildCB = [&](const float* bi, const float* bh, int tile) -> v8f {
    int mb = tile * 16 + (hi ? 8 : 0);
    v8f c;
#pragma unroll
    for (int r = 0; r < 8; ++r) {
      int m = mb + r;
      c[r] = (m < HID) ? (bi[m] + bh[m]) : 0.0f;
    }
    return c;
  };

  // Initial state as f16 B operand (lane = batch col, K = hidden idx)
  auto buildB = [&](const float* H) -> v16h {
    const float* hr = H + (size_t)(bbase + col) * HID;
    int kb = hi ? 8 : 0;
    v16h b;
#pragma unroll
    for (int e = 0; e < 16; ++e) {
      int k = (e < 8) ? (kb + e) : (16 + kb + (e - 8));
      b[e] = (_Float16)((k < HID) ? hr[k] : 0.0f);
    }
    return b;
  };

  // Initial state as f32 C-layout tile
  auto buildS = [&](const float* H, int tile) -> v8f {
    const float* hr = H + (size_t)(bbase + col) * HID;
    int mb = tile * 16 + (hi ? 8 : 0);
    v8f s;
#pragma unroll
    for (int r = 0; r < 8; ++r) {
      int m = mb + r;
      s[r] = (m < HID) ? hr[m] : 0.0f;
    }
    return s;
  };

  const v16h a0t0  = buildA(Whh0, Wih0, 0), a0t1  = buildA(Whh0, Wih0, 1);
  const v16h ah1t0 = buildA(Whh1, nullptr, 0), ah1t1 = buildA(Whh1, nullptr, 1);
  const v16h ai1t0 = buildA(Wih1, nullptr, 0), ai1t1 = buildA(Wih1, nullptr, 1);
  const v16h ah2t0 = buildA(Whh2, nullptr, 0), ah2t1 = buildA(Whh2, nullptr, 1);
  const v16h ai2t0 = buildA(Wih2, nullptr, 0), ai2t1 = buildA(Wih2, nullptr, 1);

  const v8f cb00 = buildCB(bih0, bhh0, 0), cb01 = buildCB(bih0, bhh0, 1);
  const v8f cb10 = buildCB(bih1, bhh1, 0), cb11 = buildCB(bih1, bhh1, 1);
  const v8f cb20 = buildCB(bih2, bhh2, 0), cb21 = buildCB(bih2, bhh2, 1);

  // recurrent state: f16 B operands (feed WMMA).
  // s** hold PRE-ACTIVATION f32 of the most recent step (relu applied at end).
  v16h sb0 = buildB(hin);
  v16h sb1 = buildB(hin + (size_t)B * HID);
  v16h sb2 = buildB(hin + (size_t)2 * B * HID);
  v8f s00 = buildS(hin, 0),                      s01 = buildS(hin, 1);
  v8f s10 = buildS(hin + (size_t)B * HID, 0),    s11 = buildS(hin + (size_t)B * HID, 1);
  v8f s20 = buildS(hin + (size_t)2 * B * HID,0), s21 = buildS(hin + (size_t)2 * B * HID, 1);
  // note: initial h is already post-activation (>=0), so epilogue relu is a no-op on it

  // ---------------- recurrence: T steps, 4x unrolled, x prefetched ----------
  const float* xrow = x + (size_t)(bbase + col) * T;
  float4 xq = *(const float4*)(xrow);
  for (int t = 0; t < T; t += 4) {
    // long-range cacheline prefetch (counter-free global_prefetch_b8):
    // 64 elements = 16 iterations ahead, so the demand load below always hits.
    int tp = (t + 64 < T) ? (t + 64) : 0;
    __builtin_prefetch(xrow + tp, 0, 1);

    int tn = (t + 4 < T) ? (t + 4) : 0;              // wrap: harmless dummy
    float4 xq_next = *(const float4*)(xrow + tn);
    float xs[4] = {xq.x, xq.y, xq.z, xq.w};
#pragma unroll
    for (int u = 0; u < 4; ++u) {
      // ---- layer 0: inject x_t as K=24 column (lane>=16, element 8) ----
      v16h bb = sb0;
      _Float16 xh = (_Float16)xs[u];
      bb[8] = hi ? xh : bb[8];                       // v_cndmask_b16, no divergence
      s00 = wmma_f16(a0t0, bb, cb00);
      s01 = wmma_f16(a0t1, bb, cb01);
      sb0 = pack_relu(s00, s01);

      // ---- layer 1: W_hh1*h1 + W_ih1*h0_new + b ----
      v8f q0 = wmma_f16(ah1t0, sb1, cb10);
      v8f q1 = wmma_f16(ah1t1, sb1, cb11);
      s10 = wmma_f16(ai1t0, sb0, q0);
      s11 = wmma_f16(ai1t1, sb0, q1);
      sb1 = pack_relu(s10, s11);

      // ---- layer 2: W_hh2*h2 + W_ih2*h1_new + b ----
      v8f r0 = wmma_f16(ah2t0, sb2, cb20);
      v8f r1 = wmma_f16(ah2t1, sb2, cb21);
      s20 = wmma_f16(ai2t0, sb1, r0);
      s21 = wmma_f16(ai2t1, sb1, r1);
      sb2 = pack_relu(s20, s21);
    }
    xq = xq_next;
  }

  // epilogue relu (pre-activation -> hidden state), f32
  auto relu8 = [](v8f v) -> v8f {
    v8f o;
#pragma unroll
    for (int r = 0; r < 8; ++r) o[r] = fmaxf(v[r], 0.0f);
    return o;
  };
  s00 = relu8(s00); s01 = relu8(s01);
  s10 = relu8(s10); s11 = relu8(s11);
  s20 = relu8(s20); s21 = relu8(s21);

  // ---------------- MLP head on last step (f32, via LDS staging) ------------
  __shared__ float h2s[16][HID];
#pragma unroll
  for (int r = 0; r < 8; ++r) {
    int m0 = r + (hi ? 8 : 0);                       // 0..15
    h2s[col][m0] = s20[r];
    int m1 = 16 + r + (hi ? 8 : 0);                  // 16..31
    if (m1 < HID) h2s[col][m1] = s21[r];             // lanes<16 only
  }
  __syncthreads();

  if (lane < 16) {
    const float* hrow = h2s[lane];
    float z[HID];
#pragma unroll
    for (int n = 0; n < HID; ++n) {
      float acc = b1[n];
#pragma unroll
      for (int k = 0; k < HID; ++k) acc += W1[n * HID + k] * hrow[k];
      z[n] = fmaxf(acc, 0.0f);
    }
    float y = b2[0];
#pragma unroll
    for (int n = 0; n < HID; ++n) y += W2[n] * z[n];
    out[bbase + lane] = fmaxf(y, 0.0f);
  }

  // ---------------- final hidden states: out[B + l*B*24 + b*24 + m] ---------
  float* hout = out + B;
  auto storeS = [&](v8f st0, v8f st1, int l) {
    float* hl = hout + (size_t)l * B * HID + (size_t)(bbase + col) * HID;
#pragma unroll
    for (int r = 0; r < 8; ++r) {
      int m0 = r + (hi ? 8 : 0);
      hl[m0] = st0[r];
      int m1 = 16 + r + (hi ? 8 : 0);
      if (m1 < HID) hl[m1] = st1[r];
    }
  };
  storeS(s00, s01, 0);
  storeS(s10, s11, 1);
  storeS(s20, s21, 2);
}

extern "C" void kernel_launch(void* const* d_in, const int* in_sizes, int n_in,
                              void* d_out, int out_size, void* d_ws, size_t ws_size,
                              hipStream_t stream) {
  (void)n_in; (void)out_size; (void)d_ws; (void)ws_size;
  const float* x    = (const float*)d_in[0];
  const float* h    = (const float*)d_in[1];
  const float* Wih0 = (const float*)d_in[2];
  const float* bih0 = (const float*)d_in[3];
  const float* Whh0 = (const float*)d_in[4];
  const float* bhh0 = (const float*)d_in[5];
  const float* Wih1 = (const float*)d_in[6];
  const float* bih1 = (const float*)d_in[7];
  const float* Whh1 = (const float*)d_in[8];
  const float* bhh1 = (const float*)d_in[9];
  const float* Wih2 = (const float*)d_in[10];
  const float* bih2 = (const float*)d_in[11];
  const float* Whh2 = (const float*)d_in[12];
  const float* bhh2 = (const float*)d_in[13];
  const float* W1   = (const float*)d_in[14];
  const float* b1   = (const float*)d_in[15];
  const float* W2   = (const float*)d_in[16];
  const float* b2   = (const float*)d_in[17];
  float* out = (float*)d_out;

  int B = in_sizes[1] / (3 * HID);        // 1024
  int T = in_sizes[0] / B;                // 4096

  dim3 grid(B / 16), block(32);
  rnn3_wmma_kernel<<<grid, block, 0, stream>>>(
      x, h, Wih0, bih0, Whh0, bhh0, Wih1, bih1, Whh1, bhh1,
      Wih2, bih2, Whh2, bhh2, W1, b1, W2, b2, out, B, T);
}
